// SelfAttention_28243704938857
// MI455X (gfx1250) — matrix operand order repaired and checked by reading, not compiled, not run on hardware
//
#include <hip/hip_runtime.h>

typedef __attribute__((ext_vector_type(16))) _Float16 v16h;
typedef __attribute__((ext_vector_type(8)))  _Float16 v8h;
typedef __attribute__((ext_vector_type(8)))  float    v8f;

#define SEQ_L   2048
#define CH      32          // per-head dim
#define S_TILE  64          // s-chunk staged in LDS
#define WAVES   8
#define T_TILE  16          // rows per WMMA tile; each wave owns 2 tiles = 32 t

// LDS row strides chosen so the WMMA-operand b128 reads hit disjoint 4-bank sets
#define KT_STRIDE 40        // halves per row of K^T[s][c]  (80B rows, 16B aligned)
#define V_STRIDE  72        // halves per row of V[c][s]    (144B rows, 16B aligned)
#define P_STRIDE  40        // halves per row of P[t][s]
#define O_STRIDE  18        // floats per row of O[c][t]

__global__ __launch_bounds__(256) void
attn_fwd_kernel(const float* __restrict__ in, float* __restrict__ out)
{
    __shared__ _Float16 ldsKT[S_TILE][KT_STRIDE];
    __shared__ _Float16 ldsV [CH][V_STRIDE];
    __shared__ _Float16 ldsP [WAVES][2][T_TILE][P_STRIDE];
    __shared__ float    ldsO [WAVES][CH][O_STRIDE];

    const int tid  = threadIdx.x;
    const int lane = tid & 31;
    const int wv   = tid >> 5;
    const int n16  = lane & 15;      // column index inside 16-lane half
    const int hi   = lane >> 4;      // which 16-lane half

    const int head = blockIdx.y;     // 0..63 = b*16 + h
    const int b    = head >> 4;
    const int h    = head & 15;

    const size_t qoff = ((size_t)b * 1536 +        h * CH) * SEQ_L;
    const size_t koff = ((size_t)b * 1536 +  512 + h * CH) * SEQ_L;
    const size_t voff = ((size_t)b * 1536 + 1024 + h * CH) * SEQ_L;

    const int t0 = (blockIdx.x * WAVES + wv) * (2 * T_TILE);

    // ---- Q tiles (2 x 16 t x 32 c) in WMMA A layout.
    // Fold softmax scale AND log2(e) in, so scores come out ready for v_exp_f32 (base-2).
    const float qscale = 0.2550352425f;   // log2(e)/sqrt(32)
    v16h qa0, qa1;
    #pragma unroll
    for (int i = 0; i < 16; ++i) {
        const int c = (hi << 4) + i;   // A layout k-map
        const float* qp = in + qoff + (size_t)c * SEQ_L + t0 + n16;
        qa0[i] = (_Float16)(qp[0]  * qscale);
        qa1[i] = (_Float16)(qp[16] * qscale);
    }

    v8f acc00, acc01, acc10, acc11;
    #pragma unroll
    for (int i = 0; i < 8; ++i) { acc00[i] = 0.f; acc01[i] = 0.f; acc10[i] = 0.f; acc11[i] = 0.f; }
    float rs0[8], rs1[8];
    #pragma unroll
    for (int i = 0; i < 8; ++i) { rs0[i] = 0.f; rs1[i] = 0.f; }

    // staging assignments
    const int sK   = tid & 63;          // K^T: this thread owns s-row sK ...
    const int cg8  = (tid >> 6) << 3;   // ... and 8 consecutive channels cg8..cg8+7
    const int cV   = tid >> 3;          // V: channel 0..31
    const int sV   = (tid & 7) << 3;    // V: s offset 0,8,...,56

    // ---- software pipeline: registers hold chunk i+1 while computing chunk i
    float  kr[8];
    float4 vr0, vr1;
    {
        #pragma unroll
        for (int j = 0; j < 8; ++j)
            kr[j] = in[koff + (size_t)(cg8 + j) * SEQ_L + sK];
        const float4* vp = (const float4*)(in + voff + (size_t)cV * SEQ_L + sV);
        vr0 = vp[0]; vr1 = vp[1];
    }

    for (int sc = 0; sc < SEQ_L; sc += S_TILE) {
        __syncthreads();   // all waves done reading the previous chunk
        // ---- commit prefetched chunk to LDS as f16 (single b128 store each)
        {
            v8h kh;
            #pragma unroll
            for (int j = 0; j < 8; ++j) kh[j] = (_Float16)kr[j];
            *(v8h*)&ldsKT[sK][cg8] = kh;
            v8h vh;
            vh[0] = (_Float16)vr0.x; vh[1] = (_Float16)vr0.y;
            vh[2] = (_Float16)vr0.z; vh[3] = (_Float16)vr0.w;
            vh[4] = (_Float16)vr1.x; vh[5] = (_Float16)vr1.y;
            vh[6] = (_Float16)vr1.z; vh[7] = (_Float16)vr1.w;
            *(v8h*)&ldsV[cV][sV] = vh;
        }
        __syncthreads();

        // ---- issue global loads for next chunk (wraps at end; harmless)
        {
            const int scn = (sc + S_TILE) & (SEQ_L - 1);
            #pragma unroll
            for (int j = 0; j < 8; ++j)
                kr[j] = in[koff + (size_t)(cg8 + j) * SEQ_L + scn + sK];
            const float4* vp = (const float4*)(in + voff + (size_t)cV * SEQ_L + scn + sV);
            vr0 = vp[0]; vr1 = vp[1];
            // warm caches two chunks ahead (global_prefetch_b8)
            const int scp = (sc + 2 * S_TILE) & (SEQ_L - 1);
            __builtin_prefetch(in + koff + (size_t)(cg8 + (sK & 7)) * SEQ_L + scp + (sK & ~7), 0, 3);
            __builtin_prefetch(in + voff + (size_t)cV * SEQ_L + scp + sV, 0, 3);
        }

        // ---- compute on chunk sc
        #pragma unroll
        for (int half = 0; half < 2; ++half) {          // 32 s per PV step (K=32)
            // scores + exp2 for two 16x16 subtiles x two t-tiles (B tile shared)
            #pragma unroll
            for (int ss = 0; ss < 2; ++ss) {
                const int srow = half * 32 + ss * 16;
                v16h b1 = *(const v16h*)&ldsKT[srow + n16][hi << 4];
                v8f z;
                #pragma unroll
                for (int i = 0; i < 8; ++i) z[i] = 0.f;
                v8f sv0 = __builtin_amdgcn_wmma_f32_16x16x32_f16(
                              false, qa0, false, b1, (short)0, z, false, false);
                v8f sv1 = __builtin_amdgcn_wmma_f32_16x16x32_f16(
                              false, qa1, false, b1, (short)0, z, false, false);
                #pragma unroll
                for (int v = 0; v < 8; ++v) {
                    float p0 = __builtin_amdgcn_exp2f(sv0[v]);
                    float p1 = __builtin_amdgcn_exp2f(sv1[v]);
                    rs0[v] += p0;
                    rs1[v] += p1;
                    ldsP[wv][0][v + (hi << 3)][n16 + (ss << 4)] = (_Float16)p0;
                    ldsP[wv][1][v + (hi << 3)][n16 + (ss << 4)] = (_Float16)p1;
                }
            }
            asm volatile("s_wait_dscnt 0" ::: "memory");   // P stores visible to whole wave
            // P tiles in A layout; shared V columns in B layout; two 16-wide c blocks
            v16h b20 = *(const v16h*)&ldsV[n16     ][half * 32 + (hi << 4)];
            v16h b21 = *(const v16h*)&ldsV[n16 + 16][half * 32 + (hi << 4)];
            v8h plo0 = *(const v8h*)&ldsP[wv][0][n16][hi << 4];
            v8h phi0 = *(const v8h*)&ldsP[wv][0][n16][(hi << 4) + 8];
            v8h plo1 = *(const v8h*)&ldsP[wv][1][n16][hi << 4];
            v8h phi1 = *(const v8h*)&ldsP[wv][1][n16][(hi << 4) + 8];
            v16h a20, a21;
            #pragma unroll
            for (int i = 0; i < 8; ++i) {
                a20[i] = plo0[i]; a20[8 + i] = phi0[i];
                a21[i] = plo1[i]; a21[8 + i] = phi1[i];
            }
            acc00 = __builtin_amdgcn_wmma_f32_16x16x32_f16(
                        false, a20, false, b20, (short)0, acc00, false, false);
            acc01 = __builtin_amdgcn_wmma_f32_16x16x32_f16(
                        false, a20, false, b21, (short)0, acc01, false, false);
            acc10 = __builtin_amdgcn_wmma_f32_16x16x32_f16(
                        false, a21, false, b20, (short)0, acc10, false, false);
            acc11 = __builtin_amdgcn_wmma_f32_16x16x32_f16(
                        false, a21, false, b21, (short)0, acc11, false, false);
        }
    }

    // ---- epilogue per t-tile: row-sum reduce, normalize, LDS transpose, coalesced store
    #pragma unroll
    for (int tile = 0; tile < 2; ++tile) {
        float* rs = tile ? rs1 : rs0;
        v8f aA = tile ? acc10 : acc00;
        v8f aB = tile ? acc11 : acc01;
        #pragma unroll
        for (int v = 0; v < 8; ++v) {
            float r = rs[v];
            r += __shfl_xor(r, 1, 32);
            r += __shfl_xor(r, 2, 32);
            r += __shfl_xor(r, 4, 32);
            r += __shfl_xor(r, 8, 32);
            rs[v] = r;
        }
        #pragma unroll
        for (int v = 0; v < 8; ++v) {
            const float invr = __builtin_amdgcn_rcpf(rs[v]);
            ldsO[wv][n16     ][v + (hi << 3)] = aA[v] * invr;
            ldsO[wv][n16 + 16][v + (hi << 3)] = aB[v] * invr;
        }
        asm volatile("s_wait_dscnt 0" ::: "memory");
        #pragma unroll
        for (int it = 0; it < 16; ++it) {
            const int c = (it << 1) + hi;
            out[((size_t)b * 512 + h * CH + c) * SEQ_L + t0 + tile * 16 + n16] = ldsO[wv][c][n16];
        }
        asm volatile("s_wait_dscnt 0" ::: "memory");   // tile0 reads done before tile1 overwrites
    }
}

extern "C" void kernel_launch(void* const* d_in, const int* in_sizes, int n_in,
                              void* d_out, int out_size, void* d_ws, size_t ws_size,
                              hipStream_t stream) {
    (void)in_sizes; (void)n_in; (void)out_size; (void)d_ws; (void)ws_size;
    const float* in = (const float*)d_in[0];
    float* out = (float*)d_out;
    dim3 grid(SEQ_L / (WAVES * 2 * T_TILE), 64);   // (8 t-chunks, 64 heads)
    attn_fwd_kernel<<<grid, 256, 0, stream>>>(in, out);
}
